// Head_5068061409811
// MI455X (gfx1250) — matrix operand order repaired and checked by reading, not compiled, not run on hardware
//
#include <hip/hip_runtime.h>

#define B_   16
#define T_   2048
#define C_   512
#define H_   64
#define ENDTOK 32000
#define TPAD 40                      // LDS row stride (bf16 elems), 80B = 5*16B aligned
#define QKV_BYTES ((size_t)B_ * T_ * H_ * 2)   // 4 MB per matrix in bf16

typedef __attribute__((ext_vector_type(16))) __bf16 v16bf;
typedef __attribute__((ext_vector_type(8)))  float  v8f;

union FragU { v16bf v; uint4 u[2]; };

// B-matrix fragment (32x16, contraction-major contiguous): 16 contiguous bf16 per lane
__device__ __forceinline__ v16bf load_b16(const __bf16* p) {
    FragU f;
    f.u[0] = *(const uint4*)(p);
    f.u[1] = *(const uint4*)(p + 8);
    return f.v;
}
// A-matrix fragment (16x32): 16 contiguous k per lane starting at kbase (kbase = hi*16)
__device__ __forceinline__ v16bf load_a16(const __bf16* rowp, int kbase) {
    FragU f;
    f.u[0] = *(const uint4*)(rowp + kbase);
    f.u[1] = *(const uint4*)(rowp + kbase + 8);
    return f.v;
}

__device__ __forceinline__ v8f wmma_bf16(v16bf a, v16bf b, v8f c) {
    return __builtin_amdgcn_wmma_f32_16x16x32_bf16(false, a, false, b, (short)0, c,
                                                   false, false);
}

__device__ __forceinline__ float redmax16(float x) {
    x = fmaxf(x, __shfl_xor(x, 1, 32));
    x = fmaxf(x, __shfl_xor(x, 2, 32));
    x = fmaxf(x, __shfl_xor(x, 4, 32));
    x = fmaxf(x, __shfl_xor(x, 8, 32));
    return x;
}
__device__ __forceinline__ float redsum16(float x) {
    x += __shfl_xor(x, 1, 32);
    x += __shfl_xor(x, 2, 32);
    x += __shfl_xor(x, 4, 32);
    x += __shfl_xor(x, 8, 32);
    return x;
}

// ---------------- kernel 1: first END_TOKEN position per batch ----------------
__global__ void limits_kernel(const int* __restrict__ idx, int* __restrict__ limits) {
    __shared__ int lmin;
    if (threadIdx.x == 0) lmin = T_;
    __syncthreads();
    const int b = blockIdx.x;
    int local = T_;
    for (int t = threadIdx.x; t < T_; t += blockDim.x)
        if (idx[b * T_ + t] == ENDTOK) local = min(local, t);
    atomicMin(&lmin, local);
    __syncthreads();
    if (threadIdx.x == 0) limits[b] = lmin;
}

// ---------------- kernel 2: QKV projection (WMMA bf16) ----------------
// grid 256 x 256 threads: block = 128 rows, wave = 16 rows. K = 512 in steps of 32.
__global__ __launch_bounds__(256) void qkv_kernel(
    const float* __restrict__ x, const float* __restrict__ Wk,
    const float* __restrict__ Wq, const float* __restrict__ Wv,
    __bf16* __restrict__ qs, __bf16* __restrict__ ks, __bf16* __restrict__ vt) {
    __shared__ __bf16 wt[3][H_][TPAD];   // W transposed: [matrix][n][kk]
    __shared__ __bf16 at[8][16][TPAD];   // per-wave x tile: [wave][row][kk]

    const int tid = threadIdx.x;
    const int wv = tid >> 5, lane = tid & 31;
    const int laneLo = lane & 15, hi = lane >> 4;
    const int rbase = blockIdx.x * 128 + wv * 16;

    v8f acc[12] = {};                    // [m*4 + ntile], m: 0=K 1=Q 2=V

    for (int k0 = 0; k0 < C_; k0 += 32) {
        // stage W tiles (coalesced: consecutive tid -> consecutive n)
        for (int e = tid; e < H_ * 32; e += 256) {
            const int n = e & 63, kk = e >> 6;
            const size_t gi = (size_t)(k0 + kk) * H_ + n;
            wt[0][n][kk] = (__bf16)Wk[gi];
            wt[1][n][kk] = (__bf16)Wq[gi];
            wt[2][n][kk] = (__bf16)Wv[gi];
        }
        // stage this wave's 16x32 x-tile (coalesced 128B rows)
        for (int i = 0; i < 16; i++)
            at[wv][i][lane] = (__bf16)x[(size_t)(rbase + i) * C_ + k0 + lane];
        __syncthreads();

        const v16bf afrag = load_a16(&at[wv][laneLo][0], hi * 16);
        // batch all 12 B-fragment LDS loads, then 12 back-to-back WMMAs
        v16bf bfr[12];
#pragma unroll
        for (int m = 0; m < 3; m++)
#pragma unroll
            for (int nt = 0; nt < 4; nt++)
                bfr[m * 4 + nt] = load_b16(&wt[m][nt * 16 + laneLo][hi * 16]);
#pragma unroll
        for (int i = 0; i < 12; i++)
            acc[i] = wmma_bf16(afrag, bfr[i], acc[i]);
        __syncthreads();
    }

    const float scale = 0.044194173824159216f;   // 1/sqrt(512) = C^-0.5
#pragma unroll
    for (int nt = 0; nt < 4; nt++)
#pragma unroll
        for (int r = 0; r < 8; r++) {
            const int row = rbase + r + hi * 8;
            const int h = nt * 16 + laneLo;
            ks[(size_t)row * H_ + h] = (__bf16)acc[0 * 4 + nt][r];
            qs[(size_t)row * H_ + h] = (__bf16)(acc[1 * 4 + nt][r] * scale);
            const int b = row >> 11, t = row & (T_ - 1);
            vt[(((size_t)(b * H_ + h)) << 11) + t] = (__bf16)acc[2 * 4 + nt][r];
        }
}

// ---------------- kernel 3: flash attention (WMMA bf16, online softmax) --------
// one wave per 16-query tile; 8 waves/block; grid 256.
__global__ __launch_bounds__(256) void attn_kernel(
    const __bf16* __restrict__ qs, const __bf16* __restrict__ ks,
    const __bf16* __restrict__ vt, const int* __restrict__ limits,
    float* __restrict__ out) {
    __shared__ __bf16 pstage[8][16][TPAD];       // per-wave P staging (D->A layout)

    const int tid = threadIdx.x;
    const int wv = tid >> 5, lane = tid & 31;
    const int laneLo = lane & 15, hi = lane >> 4;
    const int gw = blockIdx.x * 8 + wv;
    const int b = gw >> 7;                        // 128 q-tiles per batch
    const int qbase = (gw & 127) * 16;
    const int limit = limits[b];

    // Q A-fragments (16 rows x 64 H -> two 16x32 fragments), loaded once
    const __bf16* qrow = qs + ((size_t)(b * T_ + qbase + laneLo)) * H_;
    const v16bf qf0 = load_a16(qrow, hi * 16);
    const v16bf qf1 = load_a16(qrow + 32, hi * 16);

    float m[8], l[8];
    v8f o[4] = {};
#pragma unroll
    for (int r = 0; r < 8; r++) { m[r] = -__builtin_inff(); l[r] = 0.f; }

    const __bf16* kbase_p = ks + ((size_t)(b * T_)) * H_;
    const __bf16* vbase_p = vt + (((size_t)(b * H_)) << 11);

    const int qmax = qbase + 15;
    for (int kb = 0; kb <= qmax; kb += 32) {      // kb+31 <= 2047 always
        // ---- scores for 32 keys = two 16x16 tiles (batch loads, then WMMAs) ----
        const __bf16* kr  = kbase_p + (size_t)(kb + laneLo) * H_ + hi * 16;
        const __bf16* kr2 = kr + 16 * H_;
        const v16bf kf00 = load_b16(kr);
        const v16bf kf01 = load_b16(kr + 32);
        const v16bf kf10 = load_b16(kr2);
        const v16bf kf11 = load_b16(kr2 + 32);
        // prefetch next key-block K rows (gfx1250 global_prefetch)
        __builtin_prefetch(kr + 32 * H_, 0, 0);
        v8f sA = {}, sB = {};
        sA = wmma_bf16(qf0, kf00, sA);
        sA = wmma_bf16(qf1, kf01, sA);
        sB = wmma_bf16(qf0, kf10, sB);
        sB = wmma_bf16(qf1, kf11, sB);

        // ---- causal mask only on diagonal blocks (wave-uniform branch) ----
        if (kb + 31 > qbase) {
            const int key0 = kb + laneLo;
#pragma unroll
            for (int r = 0; r < 8; r++) {
                const int q = qbase + r + hi * 8;
                sA[r] = (key0      <= q) ? sA[r] : -__builtin_inff();
                sB[r] = (key0 + 16 <= q) ? sB[r] : -__builtin_inff();
            }
        }

        // ---- online softmax update (single fused row-max reduction) ----
        float pA[8], pB[8];
#pragma unroll
        for (int r = 0; r < 8; r++) {
            const float tmax = redmax16(fmaxf(sA[r], sB[r]));
            const float mnew = fmaxf(m[r], tmax);
            const float fr = __expf(m[r] - mnew);      // exp(-inf - finite) = 0
            m[r] = mnew;
            const float a  = __expf(sA[r] - mnew);     // masked -> exp(-inf) = 0
            const float bb = __expf(sB[r] - mnew);
            pA[r] = a; pB[r] = bb;
            l[r] = l[r] * fr + redsum16(a + bb);
#pragma unroll
            for (int t = 0; t < 4; t++) o[t][r] *= fr;
        }

        // ---- stage P (D layout) into LDS, reload as 16x32 A fragment ----
#pragma unroll
        for (int r = 0; r < 8; r++) {
            const int row = r + hi * 8;
            pstage[wv][row][laneLo]      = (__bf16)pA[r];
            pstage[wv][row][laneLo + 16] = (__bf16)pB[r];
        }
        const v16bf pf = load_a16(&pstage[wv][laneLo][0], hi * 16);  // LDS in-order

        // ---- O += P @ V  (V transposed: contiguous key-dim per h) ----
        const __bf16* vb = vbase_p + kb + hi * 16;
        const v16bf vf0 = load_b16(vb + (((size_t)(0 * 16 + laneLo)) << 11));
        const v16bf vf1 = load_b16(vb + (((size_t)(1 * 16 + laneLo)) << 11));
        const v16bf vf2 = load_b16(vb + (((size_t)(2 * 16 + laneLo)) << 11));
        const v16bf vf3 = load_b16(vb + (((size_t)(3 * 16 + laneLo)) << 11));
        __builtin_prefetch(vb + 32 + (((size_t)laneLo) << 11), 0, 0);
        o[0] = wmma_bf16(pf, vf0, o[0]);
        o[1] = wmma_bf16(pf, vf1, o[1]);
        o[2] = wmma_bf16(pf, vf2, o[2]);
        o[3] = wmma_bf16(pf, vf3, o[3]);
    }

    // ---- epilogue: normalize, end-token rows -> NaN (matches jax softmax) ----
    float* ob = out + ((size_t)(b * T_ + qbase)) * H_;
#pragma unroll
    for (int r = 0; r < 8; r++) {
        const int row = r + hi * 8;
        const bool dead = (qbase + row) >= limit;
        const float inv = 1.0f / l[r];
#pragma unroll
        for (int t = 0; t < 4; t++) {
            float val = o[t][r] * inv;
            if (dead) val = __builtin_nanf("");
            ob[(size_t)row * H_ + t * 16 + laneLo] = val;
        }
    }
}

extern "C" void kernel_launch(void* const* d_in, const int* in_sizes, int n_in,
                              void* d_out, int out_size, void* d_ws, size_t ws_size,
                              hipStream_t stream) {
    const float* x  = (const float*)d_in[0];
    const float* Wk = (const float*)d_in[1];
    const float* Wq = (const float*)d_in[2];
    const float* Wv = (const float*)d_in[3];
    const int*   idx = (const int*)d_in[4];
    float* out = (float*)d_out;

    char* ws = (char*)d_ws;
    int*    limits = (int*)ws;
    __bf16* qs = (__bf16*)(ws + 4096);
    __bf16* ks = (__bf16*)(ws + 4096 + QKV_BYTES);
    __bf16* vt = (__bf16*)(ws + 4096 + 2 * QKV_BYTES);

    limits_kernel<<<B_, 256, 0, stream>>>(idx, limits);
    qkv_kernel<<<(B_ * T_) / 128, 256, 0, stream>>>(x, Wk, Wq, Wv, qs, ks, vt);
    attn_kernel<<<(B_ * T_ / 16) / 8, 256, 0, stream>>>(qs, ks, vt, limits, out);
}
